// Embed2GraphByLinear_56307021250983
// MI455X (gfx1250) — matrix operand order repaired and checked by reading, not compiled, not run on hardware
//
#include <hip/hip_runtime.h>
#include <hip/hip_bf16.h>

// Problem constants (from reference setup_inputs)
constexpr int Bc = 8;     // batch
constexpr int Nc = 360;   // roi_num
constexpr int Dc = 64;    // input_dim
constexpr int ROWS = Bc * Nc;  // 2880, divisible by 16

typedef __attribute__((ext_vector_type(16))) _Float16 v16h;
typedef __attribute__((ext_vector_type(8)))  float    v8f;

__device__ inline void put8(v16h& dst, const float4 lo, const float4 hi, const int base) {
  dst[base + 0] = (_Float16)lo.x; dst[base + 1] = (_Float16)lo.y;
  dst[base + 2] = (_Float16)lo.z; dst[base + 3] = (_Float16)lo.w;
  dst[base + 4] = (_Float16)hi.x; dst[base + 5] = (_Float16)hi.y;
  dst[base + 6] = (_Float16)hi.z; dst[base + 7] = (_Float16)hi.w;
}

// A fragment (16x32 f16, MxK): lane holds row M=laneLo, elems 0..15 = K = kbase..kbase+15,
// where kbase = kc*32 + laneHi*16.
__device__ inline v16h load_a_frag(const float* __restrict__ xrow, const int kbase) {
  float4 f0 = *(const float4*)(xrow + kbase + 0);
  float4 f1 = *(const float4*)(xrow + kbase + 4);
  float4 f2 = *(const float4*)(xrow + kbase + 8);
  float4 f3 = *(const float4*)(xrow + kbase + 12);
  v16h a;
  put8(a, f0, f1, 0);
  put8(a, f2, f3, 8);
  return a;
}

// B fragment (32x16 f16, KxN): lane holds column N = laneLo (= one W1 row),
// elems 0..15 = 16 consecutive K values starting at c0 = colBase + kc*32 + laneHi*16.
__device__ inline v16h load_b_frag(const float* __restrict__ w1row, const int c0) {
  float4 f0 = *(const float4*)(w1row + c0 + 0);
  float4 f1 = *(const float4*)(w1row + c0 + 4);
  float4 f2 = *(const float4*)(w1row + c0 + 8);
  float4 f3 = *(const float4*)(w1row + c0 + 12);
  v16h b;
  put8(b, f0, f1, 0);
  put8(b, f2, f3, 8);
  return b;
}

// Stage 1: S = X * W1[:, :64]^T ; R = X * W1[:, 64:]^T   (X: [2880,64] f32)
// One wave per 16-row tile. S[m,o] = sum_k X[m,k]*W1[o,k]  => B[k][n] = W1[n][k] (row-contig loads).
__global__ void __launch_bounds__(32) stage1_wmma(const float* __restrict__ x,
                                                  const float* __restrict__ W1,
                                                  float* __restrict__ S,
                                                  float* __restrict__ R) {
  const int lane   = threadIdx.x;
  const int laneLo = lane & 15;
  const int laneHi = lane >> 4;
  const int row0   = blockIdx.x * 16;

  const float* xrow = x + (row0 + laneLo) * Dc;
  const v16h a0 = load_a_frag(xrow, 0 * 32 + laneHi * 16);   // K-chunk 0 (K=0..31)
  const v16h a1 = load_a_frag(xrow, 1 * 32 + laneHi * 16);   // K-chunk 1 (K=32..63)

#pragma unroll
  for (int ot = 0; ot < 4; ++ot) {  // output-feature tiles of 16
    const float* w1row = W1 + (ot * 16 + laneLo) * (2 * Dc);
    const v16h bs0 = load_b_frag(w1row, 0 * 32 + laneHi * 16);        // sender half
    const v16h bs1 = load_b_frag(w1row, 1 * 32 + laneHi * 16);
    const v16h br0 = load_b_frag(w1row, Dc + 0 * 32 + laneHi * 16);   // receiver half
    const v16h br1 = load_b_frag(w1row, Dc + 1 * 32 + laneHi * 16);

    v8f cs = {};
    v8f cr = {};
    cs = __builtin_amdgcn_wmma_f32_16x16x32_f16(false, a0, false, bs0, (short)0, cs, false, false);
    cs = __builtin_amdgcn_wmma_f32_16x16x32_f16(false, a1, false, bs1, (short)0, cs, false, false);
    cr = __builtin_amdgcn_wmma_f32_16x16x32_f16(false, a0, false, br0, (short)0, cr, false, false);
    cr = __builtin_amdgcn_wmma_f32_16x16x32_f16(false, a1, false, br1, (short)0, cr, false, false);

    // C/D layout: lane = N column (laneLo), VGPR v = M row (v + laneHi*8)
#pragma unroll
    for (int v = 0; v < 8; ++v) {
      const int m = v + laneHi * 8;
      S[(row0 + m) * Dc + ot * 16 + laneLo] = cs[v];
      R[(row0 + m) * Dc + ot * 16 + laneLo] = cr[v];
    }
  }
}

// Stage 2: m[b,i,j] = relu( sum_o relu(S[b,i,o] + R[b,j,o] + b1[o]) * w2[o] + b2 )
// with diagonal (i==j) masked: pre = b1 only.
__global__ void __launch_bounds__(256) stage2_edges(const float* __restrict__ S,
                                                    const float* __restrict__ R,
                                                    const float* __restrict__ b1,
                                                    const float* __restrict__ W2,
                                                    const float* __restrict__ b2,
                                                    float* __restrict__ out) {
  constexpr int STRIDE = 68;  // pad 64->68 words: distinct banks for the 16 j-rows
  __shared__ float sT[16 * STRIDE];
  __shared__ float rT[16 * STRIDE];
  __shared__ float b1s[64];
  __shared__ float w2s[64];

  const int t  = threadIdx.x;
  const int b  = blockIdx.y;
  const int i0 = blockIdx.x * 16;
  const int j0 = blockIdx.z * 16;

  if (t < 16) {
    ((float4*)b1s)[t] = ((const float4*)b1)[t];
  } else if (t < 32) {
    ((float4*)w2s)[t - 16] = ((const float4*)W2)[t - 16];
  }
  {
    const int row = t >> 4;         // 0..15
    const int c4  = (t & 15) * 4;   // 0,4,...,60
    const int gi = i0 + row;
    float4 sv = make_float4(0.f, 0.f, 0.f, 0.f);
    if (gi < Nc) sv = *(const float4*)(S + ((size_t)(b * Nc + gi) * Dc + c4));
    *(float4*)(sT + row * STRIDE + c4) = sv;
    const int gj = j0 + row;
    float4 rv = make_float4(0.f, 0.f, 0.f, 0.f);
    if (gj < Nc) rv = *(const float4*)(R + ((size_t)(b * Nc + gj) * Dc + c4));
    *(float4*)(rT + row * STRIDE + c4) = rv;
  }
  __syncthreads();

  const int ti = t >> 4, tj = t & 15;
  const int i = i0 + ti, j = j0 + tj;
  if (i >= Nc || j >= Nc) return;

  float acc = 0.f;
  if (i == j) {
#pragma unroll
    for (int o = 0; o < Dc; o += 4) {
      float4 bb = *(const float4*)(b1s + o);
      float4 w4 = *(const float4*)(w2s + o);
      acc += fmaxf(bb.x, 0.f) * w4.x;
      acc += fmaxf(bb.y, 0.f) * w4.y;
      acc += fmaxf(bb.z, 0.f) * w4.z;
      acc += fmaxf(bb.w, 0.f) * w4.w;
    }
  } else {
    const float* sp = sT + ti * STRIDE;
    const float* rp = rT + tj * STRIDE;
#pragma unroll
    for (int o = 0; o < Dc; o += 4) {
      float4 s4 = *(const float4*)(sp + o);
      float4 r4 = *(const float4*)(rp + o);
      float4 bb = *(const float4*)(b1s + o);
      float4 w4 = *(const float4*)(w2s + o);
      acc += fmaxf(s4.x + r4.x + bb.x, 0.f) * w4.x;
      acc += fmaxf(s4.y + r4.y + bb.y, 0.f) * w4.y;
      acc += fmaxf(s4.z + r4.z + bb.z, 0.f) * w4.z;
      acc += fmaxf(s4.w + r4.w + bb.w, 0.f) * w4.w;
    }
  }
  out[(size_t)(b * Nc + i) * Nc + j] = fmaxf(acc + b2[0], 0.f);
}

extern "C" void kernel_launch(void* const* d_in, const int* in_sizes, int n_in,
                              void* d_out, int out_size, void* d_ws, size_t ws_size,
                              hipStream_t stream) {
  const float* x  = (const float*)d_in[0];  // [8,360,64]
  const float* W1 = (const float*)d_in[1];  // [64,128]
  const float* b1 = (const float*)d_in[2];  // [64]
  const float* W2 = (const float*)d_in[3];  // [1,64]
  const float* b2 = (const float*)d_in[4];  // [1]
  float* out = (float*)d_out;               // [8,360,360,1]

  float* S = (float*)d_ws;                  // [2880,64]
  float* R = S + (size_t)ROWS * Dc;         // [2880,64]  (total 1.47 MB of ws)

  stage1_wmma<<<dim3(ROWS / 16), dim3(32), 0, stream>>>(x, W1, S, R);

  dim3 grid2((Nc + 15) / 16, Bc, (Nc + 15) / 16);  // 23 x 8 x 23
  stage2_edges<<<grid2, dim3(256), 0, stream>>>(S, R, b1, W2, b2, out);
}
